// GgmlEncoderAttention_79345225826342
// MI455X (gfx1250) — matrix operand order repaired and checked by reading, not compiled, not run on hardware
//
#include <hip/hip_runtime.h>
#include <hip/hip_bf16.h>

// ---------------------------------------------------------------------------
// Problem constants (B=4, T=2048, H=16, D=128, ATTN=2048)
// ---------------------------------------------------------------------------
#define BB 4
#define TT 2048
#define HH 16
#define DD 128
#define CC 2048                 // ATTN = H*D
#define MM (BB * TT)            // 8192 rows for all GEMMs

typedef __attribute__((ext_vector_type(16))) __bf16 v16bf;
typedef __attribute__((ext_vector_type(8)))  float  v8f;

// ---------------------------------------------------------------------------
// CDNA5 async global->LDS path (ASYNCcnt). Probe-confirmed present on this
// toolchain; prototype (from the round-2 diagnostic):
//   void __builtin_amdgcn_global_load_async_to_lds_b128(
//       v4i __attribute__((address_space(1)))* src,   // printed as __device__
//       v4i __attribute__((address_space(3)))* dst,   // LDS
//       imm int offset, imm int cpol)
// ---------------------------------------------------------------------------
#if defined(__HIP_DEVICE_COMPILE__) &&                                        \
    __has_builtin(__builtin_amdgcn_global_load_async_to_lds_b128) &&          \
    __has_builtin(__builtin_amdgcn_s_wait_asynccnt)
#define USE_ASYNC_LDS 1
#else
#define USE_ASYNC_LDS 0
#endif

#if USE_ASYNC_LDS
typedef int v4i_ __attribute__((vector_size(16)));
typedef __attribute__((address_space(1))) v4i_ as1_v4i;
typedef __attribute__((address_space(3))) v4i_ as3_v4i;
#define ASYNC_WAIT(n) __builtin_amdgcn_s_wait_asynccnt(n)
#else
#define ASYNC_WAIT(n)
#endif

// Copy one 16B chunk global -> LDS (async engine if available).
__device__ __forceinline__ void cp16_g2l(const __bf16* g, __bf16* l) {
#if USE_ASYNC_LDS
    __builtin_amdgcn_global_load_async_to_lds_b128(
        (as1_v4i*)g, (as3_v4i*)l, 0, 0);
#else
    *reinterpret_cast<uint4*>(l) = *reinterpret_cast<const uint4*>(g);
#endif
}

// ---------------------------------------------------------------------------
// Helpers
// ---------------------------------------------------------------------------
union BF16Frag { v16bf v; uint4 q[2]; };

// Build a 16x bf16 WMMA operand from two 16B chunks (LDS or global).
__device__ __forceinline__ v16bf load_frag(const __bf16* p0, const __bf16* p1) {
    BF16Frag f;
    f.q[0] = *reinterpret_cast<const uint4*>(p0);
    f.q[1] = *reinterpret_cast<const uint4*>(p1);
    return f.v;
}

// 16 f32 -> 16 bf16, stored as two 16B chunks.
__device__ __forceinline__ void cvt16_f32_bf16(const float4* src, __bf16* dst) {
    float4 a = src[0], b = src[1], c = src[2], d = src[3];
    union { __bf16 h[16]; uint4 q[2]; } t;
    t.h[0] = (__bf16)a.x;  t.h[1] = (__bf16)a.y;  t.h[2] = (__bf16)a.z;  t.h[3] = (__bf16)a.w;
    t.h[4] = (__bf16)b.x;  t.h[5] = (__bf16)b.y;  t.h[6] = (__bf16)b.z;  t.h[7] = (__bf16)b.w;
    t.h[8] = (__bf16)c.x;  t.h[9] = (__bf16)c.y;  t.h[10] = (__bf16)c.z; t.h[11] = (__bf16)c.w;
    t.h[12] = (__bf16)d.x; t.h[13] = (__bf16)d.y; t.h[14] = (__bf16)d.z; t.h[15] = (__bf16)d.w;
    reinterpret_cast<uint4*>(dst)[0] = t.q[0];
    reinterpret_cast<uint4*>(dst)[1] = t.q[1];
}

__device__ __forceinline__ v8f wmma_bf16(v16bf a, v16bf b, v8f c) {
    return __builtin_amdgcn_wmma_f32_16x16x32_bf16(
        /*neg_a=*/false, a, /*neg_b=*/false, b,
        /*c_mod=*/(short)0, c, /*reuse_a=*/false, /*reuse_b=*/false);
}

// ---------------------------------------------------------------------------
// One-shot f32 -> bf16 conversion (x and the four weight matrices). Bandwidth
// bound and amortized: halves HBM traffic for all repeatedly-read operands.
// ---------------------------------------------------------------------------
__global__ void cvt_f32_to_bf16_kernel(const float* __restrict__ src,
                                       void* __restrict__ dst, int n16) {
    const int i = blockIdx.x * blockDim.x + threadIdx.x;
    if (i >= n16) return;
    cvt16_f32_bf16(reinterpret_cast<const float4*>(src + (size_t)i * 16),
                   (__bf16*)dst + (size_t)i * 16);
}

// ---------------------------------------------------------------------------
// GEMM:  Out[m,n] = sum_k A[m,k] * W[n,k] + bias[n]     (A, W bf16)
//   Block tile 256x128, k-step 32, double-buffered async LDS staging.
//   8 waves: 4 along M x 2 along N, each wave 64x64 = 4x4 WMMA tiles.
// ---------------------------------------------------------------------------
template <bool OUT_BF16>
__global__ __launch_bounds__(256)
void gemm_bias_kernel(const void* __restrict__ Ain, const void* __restrict__ Win,
                      const float* __restrict__ bias, void* __restrict__ Out) {
    const __bf16* A = (const __bf16*)Ain;
    const __bf16* W = (const __bf16*)Win;

    __shared__ __bf16 As[2][256][40];   // 80B row stride: 16B aligned, low-conflict
    __shared__ __bf16 Bs[2][128][40];

    const int n0   = blockIdx.x * 128;
    const int m0   = blockIdx.y * 256;
    const int tid  = threadIdx.x;
    const int lane = tid & 31;
    const int wave = tid >> 5;
    const int wm   = wave & 3;          // 4 waves along M (64 rows each)
    const int wn   = wave >> 2;         // 2 waves along N (64 cols each)
    const int lo   = lane & 15;
    const int hi   = lane >> 4;

    // staging: thread t owns A row (m0+t), 32 halves/tile; and one 16-half W chunk
    const __bf16* aSrc = A + (size_t)(m0 + tid) * CC;
    const int brow = tid >> 1, bcol = (tid & 1) * 16;
    const __bf16* wSrc = W + (size_t)(n0 + brow) * CC + bcol;

    auto stage = [&](int buf, int k0) {
#pragma unroll
        for (int c = 0; c < 4; ++c)
            cp16_g2l(aSrc + k0 + c * 8, &As[buf][tid][c * 8]);
        cp16_g2l(wSrc + k0,     &Bs[buf][brow][bcol]);
        cp16_g2l(wSrc + k0 + 8, &Bs[buf][brow][bcol + 8]);
    };

    v8f acc[4][4];
#pragma unroll
    for (int mt = 0; mt < 4; ++mt)
#pragma unroll
        for (int nt = 0; nt < 4; ++nt)
            acc[mt][nt] = {};

    constexpr int KT = CC / 32;         // 64 k-tiles
    stage(0, 0);                        // prologue

    for (int kt = 0; kt < KT; ++kt) {
        const int buf = kt & 1;
        if (kt + 1 < KT) {
            stage(buf ^ 1, (kt + 1) * 32);   // 6 async ops for the next tile
            ASYNC_WAIT(6);                   // current tile's 6 are complete
        } else {
            ASYNC_WAIT(0);
        }
        __syncthreads();

        v16bf a[4], b[4];
#pragma unroll
        for (int mt = 0; mt < 4; ++mt) {
            const int row = wm * 64 + mt * 16 + lo;
            a[mt] = load_frag(&As[buf][row][hi * 8], &As[buf][row][16 + hi * 8]);
        }
#pragma unroll
        for (int nt = 0; nt < 4; ++nt) {
            const int col = wn * 64 + nt * 16 + lo;   // N index = row of Bs
            b[nt] = load_frag(&Bs[buf][col][hi * 8], &Bs[buf][col][16 + hi * 8]);
        }
#pragma unroll
        for (int mt = 0; mt < 4; ++mt)
#pragma unroll
            for (int nt = 0; nt < 4; ++nt)
                acc[mt][nt] = wmma_bf16(a[mt], b[nt], acc[mt][nt]);
        __syncthreads();
    }

    // ---- epilogue: bias + store ----
#pragma unroll
    for (int nt = 0; nt < 4; ++nt) {
        const int col = n0 + wn * 64 + nt * 16 + lo;
        const float bv = bias[col];
#pragma unroll
        for (int mt = 0; mt < 4; ++mt) {
#pragma unroll
            for (int r = 0; r < 8; ++r) {
                const int row = m0 + wm * 64 + mt * 16 + hi * 8 + r;
                const float v = acc[mt][nt][r] + bv;
                if (OUT_BF16)
                    ((__bf16*)Out)[(size_t)row * CC + col] = (__bf16)v;
                else
                    ((float*)Out)[(size_t)row * CC + col] = v;
            }
        }
    }
}

// ---------------------------------------------------------------------------
// RoPE in place on a bf16 (B*T, H*D) buffer. Reference derives cos/sin from
// FREQ_BASE internally (the freqs_cos/sin inputs are unused there).
// ---------------------------------------------------------------------------
__global__ void rope_kernel(void* __restrict__ tptr, int total) {
    __bf16* t = (__bf16*)tptr;
    const int idx = blockIdx.x * blockDim.x + threadIdx.x;
    if (idx >= total) return;
    const int i  = idx & 63;          // D/2 = 64
    const int h  = (idx >> 6) & 15;   // H = 16
    const int bt = idx >> 10;         // B*T
    const int pos = bt & (TT - 1);
    const size_t base = (size_t)bt * CC + h * DD + 2 * i;
    const float t1 = (float)t[base];
    const float t2 = (float)t[base + 1];
    const float freq = __expf(-(float)(2 * i) * (9.210340371976184f / 128.0f));
    float s, c;
    __sincosf((float)pos * freq, &s, &c);
    t[base]     = (__bf16)(t1 * c - t2 * s);
    t[base + 1] = (__bf16)(t1 * s + t2 * c);
}

// ---------------------------------------------------------------------------
// Flash attention: block = (b, h, 128 q-rows); 8 waves x 16 q-rows each.
// 32-key tiles: S = Q K^T (WMMA bf16), online softmax (f32), O += P V (WMMA).
// K tile staged through the async LDS engine; V transposed manually.
// ---------------------------------------------------------------------------
__global__ __launch_bounds__(256)
void flash_attn_kernel(const void* __restrict__ Qp, const void* __restrict__ Kp,
                       const void* __restrict__ Vp, void* __restrict__ Yp) {
    const __bf16* Q  = (const __bf16*)Qp;
    const __bf16* Km = (const __bf16*)Kp;
    const __bf16* Vm = (const __bf16*)Vp;
    __bf16* Y        = (__bf16*)Yp;

    __shared__ __bf16 Ks[32][136];     // keys x d   (row-major)
    __shared__ __bf16 Vt[128][40];     // d x keys   (transposed V)
    __shared__ __bf16 Ps[8][16][40];   // per-wave P scratch (C-layout -> A-layout)

    const int q0   = blockIdx.x * 128;
    const int h    = blockIdx.y;
    const int b    = blockIdx.z;
    const int tid  = threadIdx.x;
    const int lane = tid & 31;
    const int wave = tid >> 5;
    const int lo   = lane & 15;
    const int hi   = lane >> 4;
    const int qrow = q0 + wave * 16;   // this wave's first q row

    // Preload this wave's Q fragments: 16 rows x 128 d = 4 A-fragments (k=32 each)
    const size_t qbase = ((size_t)b * TT + qrow + lo) * CC + h * DD;
    v16bf qf[4];
#pragma unroll
    for (int kc = 0; kc < 4; ++kc)
        qf[kc] = load_frag(Q + qbase + kc * 32 + hi * 8,
                           Q + qbase + kc * 32 + 16 + hi * 8);

    v8f o[8];
#pragma unroll
    for (int dn = 0; dn < 8; ++dn) o[dn] = {};
    float mr[8], lr[8];
#pragma unroll
    for (int r = 0; r < 8; ++r) { mr[r] = -1e30f; lr[r] = 0.0f; }

    const float scale = 0.08838834764831845f;   // 1/sqrt(128)
    const int niter = q0 / 32 + 4;              // causal: keys < q0+128

    for (int it = 0; it < niter; ++it) {
        const int kb0 = it * 32;
        // ---- stage K (async engine) and V (manual transpose) ----
        {
            const int kk = tid >> 3;          // 0..31 key within tile
            const int ds = (tid & 7) * 16;    // 0..112 d offset
            const size_t src = ((size_t)b * TT + kb0 + kk) * CC + h * DD + ds;
            cp16_g2l(Km + src,     &Ks[kk][ds]);
            cp16_g2l(Km + src + 8, &Ks[kk][ds + 8]);
            union { uint4 q[2]; __bf16 hv[16]; } vu;
            vu.q[0] = reinterpret_cast<const uint4*>(Vm + src)[0];
            vu.q[1] = reinterpret_cast<const uint4*>(Vm + src)[1];
#pragma unroll
            for (int j = 0; j < 16; ++j) Vt[ds + j][kk] = vu.hv[j];
            if (it + 1 < niter)
                __builtin_prefetch(Vm + src + 32 * CC, 0, 1);
        }
        ASYNC_WAIT(0);
        __syncthreads();

        if (kb0 <= qrow + 15) {   // not fully above the diagonal for this wave
            // ---- S = Q K^T for 32 keys (two 16-key column tiles) ----
            v8f s0 = {}, s1 = {};
#pragma unroll
            for (int kc = 0; kc < 4; ++kc) {
                v16bf bk0 = load_frag(&Ks[lo][kc * 32 + hi * 8],
                                      &Ks[lo][kc * 32 + 16 + hi * 8]);
                v16bf bk1 = load_frag(&Ks[16 + lo][kc * 32 + hi * 8],
                                      &Ks[16 + lo][kc * 32 + 16 + hi * 8]);
                s0 = wmma_bf16(qf[kc], bk0, s0);
                s1 = wmma_bf16(qf[kc], bk1, s1);
            }
#pragma unroll
            for (int r = 0; r < 8; ++r) { s0[r] *= scale; s1[r] *= scale; }

            // ---- causal mask (only on diagonal tiles) ----
            if (kb0 + 31 > qrow) {
#pragma unroll
                for (int r = 0; r < 8; ++r) {
                    const int row = qrow + hi * 8 + r;
                    if (kb0 + lo > row)      s0[r] = -1e30f;
                    if (kb0 + 16 + lo > row) s1[r] = -1e30f;
                }
            }

            // ---- online softmax (row stats replicated across 16-lane halves) ----
#pragma unroll
            for (int r = 0; r < 8; ++r) {
                float rm = fmaxf(s0[r], s1[r]);
                rm = fmaxf(rm, __shfl_xor(rm, 1, 32));
                rm = fmaxf(rm, __shfl_xor(rm, 2, 32));
                rm = fmaxf(rm, __shfl_xor(rm, 4, 32));
                rm = fmaxf(rm, __shfl_xor(rm, 8, 32));
                const float mnew = fmaxf(mr[r], rm);
                const float corr = __expf(mr[r] - mnew);
                mr[r] = mnew;
                const float p0 = __expf(s0[r] - mnew);
                const float p1 = __expf(s1[r] - mnew);
                s0[r] = p0; s1[r] = p1;
                float rs = p0 + p1;
                rs += __shfl_xor(rs, 1, 32);
                rs += __shfl_xor(rs, 2, 32);
                rs += __shfl_xor(rs, 4, 32);
                rs += __shfl_xor(rs, 8, 32);
                lr[r] = lr[r] * corr + rs;
#pragma unroll
                for (int dn = 0; dn < 8; ++dn) o[dn][r] *= corr;
            }

            // ---- P: C-layout -> bf16 A-layout via per-wave LDS scratch ----
#pragma unroll
            for (int r = 0; r < 8; ++r) {
                Ps[wave][hi * 8 + r][lo]      = (__bf16)s0[r];
                Ps[wave][hi * 8 + r][16 + lo] = (__bf16)s1[r];
            }
            v16bf pf = load_frag(&Ps[wave][lo][hi * 8], &Ps[wave][lo][16 + hi * 8]);

            // ---- O += P V (8 d-tiles of 16) ----
#pragma unroll
            for (int dn = 0; dn < 8; ++dn) {
                v16bf bv = load_frag(&Vt[dn * 16 + lo][hi * 8],
                                     &Vt[dn * 16 + lo][16 + hi * 8]);
                o[dn] = wmma_bf16(pf, bv, o[dn]);
            }
        }
        __syncthreads();
    }

    // ---- normalize and store bf16 output (b*T+row, h*D+d) ----
#pragma unroll
    for (int r = 0; r < 8; ++r) {
        const float inv = 1.0f / lr[r];
        const size_t orow = ((size_t)b * TT + qrow + hi * 8 + r) * CC + h * DD;
#pragma unroll
        for (int dn = 0; dn < 8; ++dn)
            Y[orow + dn * 16 + lo] = (__bf16)(o[dn][r] * inv);
    }
}

// ---------------------------------------------------------------------------
// Launch
// ---------------------------------------------------------------------------
extern "C" void kernel_launch(void* const* d_in, const int* in_sizes, int n_in,
                              void* d_out, int out_size, void* d_ws, size_t ws_size,
                              hipStream_t stream) {
    (void)in_sizes; (void)n_in; (void)out_size; (void)ws_size;

    const float* x    = (const float*)d_in[0];
    // d_in[1], d_in[2] = freqs_cos/sin: unused by the reference math
    const float* wq_w = (const float*)d_in[3];
    const float* wq_b = (const float*)d_in[4];
    const float* wk_w = (const float*)d_in[5];
    const float* wk_b = (const float*)d_in[6];
    const float* wv_w = (const float*)d_in[7];
    const float* wv_b = (const float*)d_in[8];
    const float* wo_w = (const float*)d_in[9];
    const float* wo_b = (const float*)d_in[10];
    float* out = (float*)d_out;

    // Workspace layout (bf16 buffers): xb | wqb wkb wvb wob | qb kb vb yb
    const size_t exb = (size_t)MM * CC;   // 16.8M elems (32 MB bf16)
    const size_t ewt = (size_t)CC * CC;   // 4.2M elems  (8 MB bf16)
    char* p = (char*)d_ws;
    void* xb  = p; p += 2 * exb;
    void* wqb = p; p += 2 * ewt;
    void* wkb = p; p += 2 * ewt;
    void* wvb = p; p += 2 * ewt;
    void* wob = p; p += 2 * ewt;
    void* qb  = p; p += 2 * exb;
    void* kb  = p; p += 2 * exb;
    void* vb  = p; p += 2 * exb;
    void* yb  = p;

    // 1) one-shot bf16 conversions
    const int nx16 = (int)(exb / 16), nw16 = (int)(ewt / 16);
    cvt_f32_to_bf16_kernel<<<(nx16 + 255) / 256, 256, 0, stream>>>(x, xb, nx16);
    cvt_f32_to_bf16_kernel<<<(nw16 + 255) / 256, 256, 0, stream>>>(wq_w, wqb, nw16);
    cvt_f32_to_bf16_kernel<<<(nw16 + 255) / 256, 256, 0, stream>>>(wk_w, wkb, nw16);
    cvt_f32_to_bf16_kernel<<<(nw16 + 255) / 256, 256, 0, stream>>>(wv_w, wvb, nw16);
    cvt_f32_to_bf16_kernel<<<(nw16 + 255) / 256, 256, 0, stream>>>(wo_w, wob, nw16);

    const dim3 ggrid(CC / 128, MM / 256);   // (16, 32)
    const dim3 gblk(256);

    // 2) Q/K/V projections
    gemm_bias_kernel<true><<<ggrid, gblk, 0, stream>>>(xb, wqb, wq_b, qb);
    gemm_bias_kernel<true><<<ggrid, gblk, 0, stream>>>(xb, wkb, wk_b, kb);
    gemm_bias_kernel<true><<<ggrid, gblk, 0, stream>>>(xb, wvb, wv_b, vb);

    // 3) RoPE on q, k
    const int rope_total = BB * TT * HH * (DD / 2);
    rope_kernel<<<(rope_total + 255) / 256, 256, 0, stream>>>(qb, rope_total);
    rope_kernel<<<(rope_total + 255) / 256, 256, 0, stream>>>(kb, rope_total);

    // 4) causal flash attention
    flash_attn_kernel<<<dim3(TT / 128, HH, BB), gblk, 0, stream>>>(qb, kb, vb, yb);

    // 5) output projection -> f32
    gemm_bias_kernel<false><<<ggrid, gblk, 0, stream>>>(yb, wob, wo_b, out);
}
